// Cluster_Attention_67448166416734
// MI455X (gfx1250) — matrix operand order repaired and checked by reading, not compiled, not run on hardware
//
#include <hip/hip_runtime.h>
#include <hip/hip_bf16.h>
#include <stdint.h>

#define FEAT 128
#define KCL  8
#define NG   10000
#define NEG_SLOPE 0.45f

typedef _Float16 f16;
typedef __attribute__((ext_vector_type(16))) _Float16 v16h;
typedef __attribute__((ext_vector_type(8)))  float    v8f;

// ---------------------------------------------------------------------------
// Kernel 0: pack W1 (f32 row-major [K=128][N=128]) into WMMA B-fragment order,
// f16. Layout: dword index ((kb*8 + nt)*32 + lane)*8 + r holds
// { W1[k][n], W1[k+1][n] } with k = kb*32 + (r>>2)*16 + (lane>>4)*8 + (r&3)*2, n = nt*16 + (lane&15).
// ---------------------------------------------------------------------------
__global__ __launch_bounds__(256) void pack_w1_kernel(const float* __restrict__ W1,
                                                      uint32_t* __restrict__ W1p) {
    for (int idx = threadIdx.x; idx < 4 * 8 * 32 * 8; idx += 256) {
        int r    = idx & 7;
        int lane = (idx >> 3) & 31;
        int nt   = (idx >> 8) & 7;
        int kb   = idx >> 11;
        int ln = lane & 15, lh = lane >> 4;
        int n = nt * 16 + ln;
        int k = kb * 32 + (r >> 2) * 16 + lh * 8 + (r & 3) * 2;
        union { f16 h[2]; uint32_t u; } cv;
        cv.h[0] = (f16)W1[(size_t)k * FEAT + n];
        cv.h[1] = (f16)W1[(size_t)(k + 1) * FEAT + n];
        W1p[idx] = cv.u;
    }
}

// ---------------------------------------------------------------------------
// Kernel 1: per-graph segment sums (bandwidth-dominant: streams 256 MB of x).
// One block (128 thr) per graph. Produces
// new_x(g,k,:) = csum(g,k,:) * cnt(g,k) / sum_k cnt(g,k)^2  as f16 row-major,
// plus float cluster counts.
// ---------------------------------------------------------------------------
__global__ __launch_bounds__(128) void cluster_sum_kernel(const float* __restrict__ x,
                                                          const int* __restrict__ cls,
                                                          f16* __restrict__ newx,
                                                          float* __restrict__ cntf,
                                                          int npg) {
    __shared__ float acc[KCL * FEAT];
    __shared__ int   cnt[KCL];
    __shared__ float ratio[KCL];

    const int g = blockIdx.x;
    const int f = threadIdx.x;            // feature column, 0..127

    #pragma unroll
    for (int k = 0; k < KCL; ++k) acc[k * FEAT + f] = 0.0f;
    if (f < KCL) cnt[f] = 0;
    __syncthreads();

    const int base = g * npg;
    for (int n = 0; n < npg; ++n) {
        int c = cls[base + n];            // uniform across block -> broadcast
        acc[c * FEAT + f] += x[(size_t)(base + n) * FEAT + f];
    }
    for (int n = f; n < npg; n += 128) atomicAdd(&cnt[cls[base + n]], 1);
    __syncthreads();

    if (f == 0) {
        float ss = 0.0f;
        #pragma unroll
        for (int k = 0; k < KCL; ++k) ss += (float)cnt[k] * (float)cnt[k];
        float inv = ss > 0.0f ? 1.0f / ss : 0.0f;
        #pragma unroll
        for (int k = 0; k < KCL; ++k) ratio[k] = (float)cnt[k] * inv;
    }
    __syncthreads();

    #pragma unroll
    for (int k = 0; k < KCL; ++k) {
        float v = acc[k * FEAT + f] * ratio[k];
        newx[((size_t)(g * KCL + k)) * FEAT + f] = (f16)v;
    }
    if (f < KCL) cntf[g * KCL + f] = (float)cnt[f];
}

// ---------------------------------------------------------------------------
// Kernel 2: a = leaky_relu(new_x @ W1 + b1) @ W2 + b2 over 80,000 rows.
// 256 threads = 8 waves; each wave owns one 16-row M-tile, all 128 N columns.
// 32x v_wmma_f32_16x16x32_f16 per wave, linearized into one 32-step pipeline
// with double-buffered B fragments so each WMMA overlaps the ds_load of the
// NEXT fragment (wait dscnt<=2 instead of dscnt==0 stalls).
// h stays in registers; h@W2 fused via per-lane partials + shfl_xor reduction.
// ---------------------------------------------------------------------------
__global__ __launch_bounds__(256) void mlp_attn_kernel(const f16* __restrict__ newx,
                                                       const uint32_t* __restrict__ W1p,
                                                       const float* __restrict__ b1,
                                                       const float* __restrict__ W2,
                                                       const float* __restrict__ b2,
                                                       float* __restrict__ aOut) {
    __shared__ uint32_t w1s[8192];        // 32 KB packed W1, B-fragment order
    for (int i = threadIdx.x; i < 8192; i += 256) w1s[i] = W1p[i];
    __syncthreads();

    const int lane = threadIdx.x & 31;
    const int wave = threadIdx.x >> 5;
    const int ln = lane & 15, lh = lane >> 4;
    const int mtile = blockIdx.x * 8 + wave;
    const int row   = mtile * 16 + ln;    // A-matrix row for this lane

    union AF { uint4 q[2]; v16h h; };

    // A fragments: 16-bit A 16x32 layout. Lane (lh,ln) needs f16 elements
    // [kb*32 + lh*8 .. +7] and [kb*32 + 16 + lh*8 .. +7] of its row -> 2x b128.
    AF A[4];
    const f16* arow = newx + (size_t)row * FEAT;
    #pragma unroll
    for (int kb = 0; kb < 4; ++kb) {
        A[kb].q[0] = *(const uint4*)(arow + kb * 32 + lh * 8);
        A[kb].q[1] = *(const uint4*)(arow + kb * 32 + 16 + lh * 8);
    }

    // Base LDS pointer for this lane's B fragments; fragment t = kb*8 + nt
    // lives at dwords (t*32 + lane)*8 .. +7  (32 bytes, 2x ds_load_b128).
    const uint4* bbase = (const uint4*)&w1s[lane * 8];
    // stride between consecutive fragments in uint4 units: 32*8 dwords = 64 uint4
    #define BFRAG(t) (bbase + (t) * 64)

    v8f acc[8];
    #pragma unroll
    for (int nt = 0; nt < 8; ++nt) acc[nt] = (v8f){};

    // 32-step software pipeline over fragments t = kb*8 + nt.
    AF Bc, Bn;
    Bc.q[0] = BFRAG(0)[0];
    Bc.q[1] = BFRAG(0)[1];
    #pragma unroll
    for (int t = 0; t < 32; ++t) {
        if (t < 31) {
            Bn.q[0] = BFRAG(t + 1)[0];
            Bn.q[1] = BFRAG(t + 1)[1];
        }
        const int kb = t >> 3;
        const int nt = t & 7;
        acc[nt] = __builtin_amdgcn_wmma_f32_16x16x32_f16(
            /*neg_a=*/false, A[kb].h, /*neg_b=*/false, Bc.h,
            /*c_mod=*/(short)0, acc[nt], /*reuse_a=*/false, /*reuse_b=*/false);
        if (t < 31) Bc = Bn;
    }
    #undef BFRAG

    // Epilogue: C/D layout => VGPR r, lanes 0-15: (M=r, N=ln);
    // lanes 16-31: (M=r+8, N=ln). Fuse bias + leaky_relu + dot with W2.
    float pa[8];
    #pragma unroll
    for (int r = 0; r < 8; ++r) pa[r] = 0.0f;

    #pragma unroll
    for (int nt = 0; nt < 8; ++nt) {
        const int n = nt * 16 + ln;
        const float bn = b1[n];
        const float wn = W2[n];
        #pragma unroll
        for (int r = 0; r < 8; ++r) {
            float h = acc[nt][r] + bn;
            h = h > 0.0f ? h : h * NEG_SLOPE;
            pa[r] += h * wn;
        }
    }

    const float bb = b2[0];
    #pragma unroll
    for (int r = 0; r < 8; ++r) {
        float s = pa[r];
        #pragma unroll
        for (int off = 8; off > 0; off >>= 1) s += __shfl_xor(s, off, 16);
        if (ln == 0) aOut[mtile * 16 + lh * 8 + r] = s + bb;
    }
}

// ---------------------------------------------------------------------------
// Kernel 3: per-graph numerically stable segment-softmax expanded back to
// nodes. Clusters with cnt==0 contribute nothing (match reference).
// ---------------------------------------------------------------------------
__global__ __launch_bounds__(64) void softmax_out_kernel(const int* __restrict__ cls,
                                                         const float* __restrict__ aArr,
                                                         const float* __restrict__ cntArr,
                                                         float* __restrict__ out,
                                                         int npg) {
    const int g = blockIdx.x;
    float m = -3.402823e38f;
    #pragma unroll
    for (int k = 0; k < KCL; ++k) {
        float c = cntArr[g * KCL + k];
        float a = aArr[g * KCL + k];
        if (c > 0.0f) m = fmaxf(m, a);
    }
    float s = 0.0f;
    #pragma unroll
    for (int k = 0; k < KCL; ++k) {
        float c = cntArr[g * KCL + k];
        float a = aArr[g * KCL + k];
        if (c > 0.0f) s += c * __expf(a - m);
    }
    const float inv = 1.0f / (s + 1e-16f);
    for (int n = threadIdx.x; n < npg; n += 64) {
        int node = g * npg + n;
        int c = cls[node];
        out[node] = __expf(aArr[g * KCL + c] - m) * inv;
    }
}

// ---------------------------------------------------------------------------
extern "C" void kernel_launch(void* const* d_in, const int* in_sizes, int n_in,
                              void* d_out, int out_size, void* d_ws, size_t ws_size,
                              hipStream_t stream) {
    const float* x   = (const float*)d_in[0];
    const int*   cls = (const int*)d_in[1];
    // d_in[2] = batch (sorted, equal-size graphs) -- not needed
    const float* W1  = (const float*)d_in[3];
    const float* b1  = (const float*)d_in[4];
    const float* W2  = (const float*)d_in[5];
    const float* b2  = (const float*)d_in[6];
    float* out = (float*)d_out;

    const int N   = in_sizes[0] / FEAT;   // 500,000
    const int G   = NG;                   // 10,000
    const int npg = N / G;                // 50 nodes per graph
    const int GK  = G * KCL;              // 80,000 rows

    // Workspace layout
    char* ws = (char*)d_ws;
    size_t off = 0;
    f16* newx = (f16*)(ws + off);
    off += (size_t)GK * FEAT * sizeof(f16);           // 20.48 MB
    off = (off + 255) & ~(size_t)255;
    uint32_t* W1p = (uint32_t*)(ws + off);
    off += 8192 * sizeof(uint32_t);                   // 32 KB
    off = (off + 255) & ~(size_t)255;
    float* aArr = (float*)(ws + off);
    off += (size_t)GK * sizeof(float);                // 320 KB
    off = (off + 255) & ~(size_t)255;
    float* cntArr = (float*)(ws + off);               // 320 KB

    pack_w1_kernel<<<1, 256, 0, stream>>>(W1, W1p);
    cluster_sum_kernel<<<G, 128, 0, stream>>>(x, cls, newx, cntArr, npg);
    mlp_attn_kernel<<<GK / 128, 256, 0, stream>>>(newx, W1p, b1, W2, b2, aArr);
    softmax_out_kernel<<<G, 64, 0, stream>>>(cls, aArr, cntArr, out, npg);
}
